// MessagePassing_40037685133518
// MI455X (gfx1250) — matrix operand order repaired and checked by reading, not compiled, not run on hardware
//
#include <hip/hip_runtime.h>
#include <math.h>

#define NW 4                 // waves per block in message kernel
#define TPB_MSG (NW * 32)

typedef __attribute__((ext_vector_type(16))) _Float16 v16h;
typedef __attribute__((ext_vector_type(8)))  float    v8f;
typedef unsigned int u32x4 __attribute__((ext_vector_type(4)));
typedef int          i32x4 __attribute__((ext_vector_type(4)));
typedef int          i32x8 __attribute__((ext_vector_type(8)));

#if defined(__has_builtin)
#if __has_builtin(__builtin_amdgcn_tensor_load_to_lds)
#define HAVE_TDM 1
#endif
#endif

union HFrag { v16h v; uint4 q[2]; _Float16 e[16]; };
union CFrag { v8f  v; float e[8]; };

static __device__ inline v16h splat16(_Float16 s) {
  v16h v;
  #pragma unroll
  for (int i = 0; i < 16; ++i) v[i] = s;
  return v;
}

// ---------------------------------------------------------------------------
// Fused weight matrix, stored DIRECTLY in WMMA A-fragment order.
// Logical matrix (row-major): Wbig[c, i], c in [0,(K+1)*64), i in [0,64):
//   Wbig[k*64+j, i] = W[k*4096 + i*64 + j]  (k < K);  = b[i*64+j]  (k == K)
// A-matrix for the WMMA is Wbig^T tiles: A[M=i_local, Kc=c_local].
// Fragment order: frag f = kkb*4 + mb; within frag: lane L (32), elem t (16):
//   i = mb*16 + (L&15)
//   c = kkb*32 + ((t>>1)<4 ? 0:16) + (L>>4)*8 + ((t>>1)&3)*2 + (t&1)
// Each lane's 16 halves are contiguous -> 2x ds_load_b128 at use time.
// ---------------------------------------------------------------------------
template<int K>
__global__ void build_wbigA_kernel(const float* __restrict__ W,
                                   const float* __restrict__ b,
                                   _Float16* __restrict__ out) {
  const int total = (K + 1) * 64 * 64;
  int t = blockIdx.x * blockDim.x + threadIdx.x;
  if (t >= total) return;
  const int frag = t >> 9;          // 512 halves per fragment
  const int rem  = t & 511;
  const int lane = rem >> 4;
  const int tt   = rem & 15;
  const int kkb  = frag >> 2;
  const int mb   = frag & 3;
  const int hl   = lane >> 4;
  const int l15  = lane & 15;
  const int vg   = tt >> 1, w = tt & 1;
  const int c = kkb * 32 + ((vg < 4) ? 0 : 16) + hl * 8 + (vg & 3) * 2 + w;
  const int i = mb * 16 + l15;
  const int k = c >> 6;
  const int j = c & 63;
  const float v = (k < K) ? W[k * 4096 + i * 64 + j] : b[i * 64 + j];
  out[t] = (_Float16)v;
}

__global__ void f32_to_f16_kernel(const float* __restrict__ in,
                                  _Float16* __restrict__ out, int n) {
  int t = blockIdx.x * blockDim.x + threadIdx.x;
  if (t < n) out[t] = (_Float16)in[t];
}

// ---------------------------------------------------------------------------
// Message phase: out[dst(e)] += z[e] @ Wbig, computed as
//   D(16 outs x 16 edges) = A(Wbig^T tile, LDS pre-swizzled) x B(z^T tile)
// Weights are DMA'd into LDS by the Tensor Data Mover when available.
// ---------------------------------------------------------------------------
template<int K>
__global__ __launch_bounds__(TPB_MSG) void msg_kernel(
    const _Float16* __restrict__ xh,     // (NA, 64) f16
    const float*    __restrict__ feat,   // (E, K)
    const int*      __restrict__ idx,    // (E, 2) [dst, src]
    const _Float16* __restrict__ WbigA,  // (K+1)*4096 halves, fragment order
    float*          __restrict__ out,    // (NA, 64) f32, pre-zeroed
    int E)
{
  constexpr int KK  = (K + 1) * 64;
  constexpr int NKB = KK / 32;
  constexpr int FP  = K + 1;            // feature row + 1.0 sentinel
  extern __shared__ char smem[];
  _Float16* sWA = (_Float16*)smem;                                    // KK*64
  _Float16* sFh = (_Float16*)(smem + (size_t)KK * 64 * 2);            // NW*16*FP
  _Float16* sX  = (_Float16*)(smem + (size_t)KK * 64 * 2
                                    + (size_t)NW * 16 * FP * 2);      // NW*1024

#if HAVE_TDM
  // ---- Tensor Data Mover: one DMA pulls the whole weight image to LDS ----
  if (threadIdx.x < 32) {               // wave 0 only (TDM ignores EXEC)
    const unsigned long long ga = (unsigned long long)(uintptr_t)WbigA;
    const unsigned int N8 = (unsigned int)(KK * 16);  // 8-byte elements
    u32x4 g0;
    g0[0] = 1u;                                        // count=1, user D#
    g0[1] = (unsigned int)(uintptr_t)sWA;              // LDS byte offset
    g0[2] = (unsigned int)ga;                          // global addr lo
    g0[3] = ((unsigned int)(ga >> 32) & 0x1FFFFFFu) | (2u << 30); // hi | type=2
    i32x8 g1;
    g1[0] = (int)(3u << 16);                           // data_size = 8B
    g1[1] = (int)((N8 & 0xFFFFu) << 16);               // tensor_dim0 lo16
    g1[2] = (int)((N8 >> 16) | (1u << 16));            // dim0 hi | tensor_dim1=1
    g1[3] = (int)((N8 & 0xFFFFu) << 16);               // tile_dim0 = N8
    g1[4] = 1;                                         // tile_dim1 = 1
    g1[5] = (int)N8;                                   // tensor_dim0_stride
    g1[6] = 0;
    g1[7] = 0;
    i32x4 gz = {0, 0, 0, 0};
#if __clang_major__ >= 23
    i32x8 gz8 = {0, 0, 0, 0, 0, 0, 0, 0};
    __builtin_amdgcn_tensor_load_to_lds(g0, g1, gz, gz, gz8, 0);
#else
    __builtin_amdgcn_tensor_load_to_lds(g0, g1, gz, gz, 0);
#endif
#if __has_builtin(__builtin_amdgcn_s_wait_tensorcnt)
    __builtin_amdgcn_s_wait_tensorcnt(0);
#else
    asm volatile("s_wait_tensorcnt 0x0" ::: "memory");
#endif
  }
#else
  // ---- fallback: cooperative b128 copy ----
  {
    const uint4* src = (const uint4*)WbigA;
    uint4*       dst = (uint4*)sWA;
    const int n16 = KK * 64 / 8;
    for (int t = threadIdx.x; t < n16; t += blockDim.x) dst[t] = src[t];
  }
#endif
  __syncthreads();

  const int wave = threadIdx.x >> 5;
  const int lane = threadIdx.x & 31;
  const int r16  = lane & 15;    // edge within tile (B column / D column)
  const int hl   = lane >> 4;
  const int tilesTotal = (E + 15) >> 4;

  for (int t0 = blockIdx.x * NW; t0 < tilesTotal; t0 += gridDim.x * NW) {
    const int tile = t0 + wave;
    if (tile >= tilesTotal) continue;

    // ---- stage: this lane's edge, gathered x row, f16 feature rows ----
    const int  e  = tile * 16 + r16;
    const bool ev = e < E;
    const int  srcAtom = ev ? idx[2 * e + 1] : 0;
    const int  dstAtom = ev ? idx[2 * e + 0] : 0;
    {
      uint4* d = (uint4*)&sX[wave * 1024 + r16 * 64 + hl * 32];
      if (ev) {
        const uint4* s = (const uint4*)(xh + (size_t)srcAtom * 64 + hl * 32);
        d[0] = s[0]; d[1] = s[1]; d[2] = s[2]; d[3] = s[3];
      } else {
        uint4 z = {0u, 0u, 0u, 0u};
        d[0] = z; d[1] = z; d[2] = z; d[3] = z;
      }
    }
    // feature rows with 1.0 sentinel at k == K (bias rows) -> branch-free load
    for (int t = lane; t < 16 * FP; t += 32) {
      const int ee = tile * 16 + t / FP;
      const int kk = t % FP;
      _Float16 v = (_Float16)1.0f;
      if (kk < K) v = (ee < E) ? (_Float16)feat[(size_t)ee * K + kk]
                               : (_Float16)0.f;
      sFh[wave * 16 * FP + t] = v;
    }
    // Same-wave LDS ops are in-order; regions are wave-private.

    CFrag acc[4];
    #pragma unroll
    for (int mb = 0; mb < 4; ++mb)
      #pragma unroll
      for (int r = 0; r < 8; ++r) acc[mb].e[r] = 0.f;

    const _Float16* xrow = &sX[wave * 1024 + r16 * 64];
    const _Float16* frow = &sFh[wave * 16 * FP + r16 * FP];

    for (int kkb = 0; kkb < NKB; ++kkb) {
      // ---- B fragment: z values k = 8h+0..7 and 16+8h+0..7 for this lane's edge ----
      const int c0   = kkb * 32 + hl * 8;
      const int kIdx = c0 >> 6;           // in [0, K]; K -> sentinel 1.0
      const int j0   = c0 & 63;
      HFrag bf;
      {
        const uint4* xs = (const uint4*)(xrow + j0);   // 16B aligned
        bf.q[0] = xs[0]; bf.q[1] = xs[2];
      }
      bf.v = bf.v * splat16(frow[kIdx]);               // v_pk_mul_f16 x8

      // ---- issue all A-fragment loads first, then retire 4 WMMAs ----
      HFrag af[4];
      #pragma unroll
      for (int mb = 0; mb < 4; ++mb) {
        const uint4* ws =
            (const uint4*)(sWA + ((size_t)(kkb * 4 + mb) * 32 + lane) * 16);
        af[mb].q[0] = ws[0]; af[mb].q[1] = ws[1];      // 2x ds_load_b128
      }
      #pragma unroll
      for (int mb = 0; mb < 4; ++mb) {
        acc[mb].v = __builtin_amdgcn_wmma_f32_16x16x32_f16(
            false, af[mb].v, false, bf.v, (short)0, acc[mb].v, false, false);
      }
    }

    // ---- scatter: lane owns edge r16; D gives outputs i = mb*16 + hl*8 + r
    if (ev) {
      float* orow = out + (size_t)dstAtom * 64;
      #pragma unroll
      for (int mb = 0; mb < 4; ++mb) {
        #pragma unroll
        for (int r = 0; r < 8; ++r) {
          atomicAdd(orow + mb * 16 + hl * 8 + r, acc[mb].e[r]);
        }
      }
    }
  }
}

// ---------------------------------------------------------------------------
// GRU cell: thread = (atom, unit). x/h rows staged in LDS, weights via L2.
// ---------------------------------------------------------------------------
__global__ __launch_bounds__(256) void gru_kernel(
    const float* __restrict__ x,  const float* __restrict__ hp,
    const float* __restrict__ wi, const float* __restrict__ wh,
    const float* __restrict__ bi, const float* __restrict__ bh,
    float* __restrict__ ho, int nAtoms)
{
  __shared__ float sx[4][64];
  __shared__ float sh[4][64];
  const int la   = threadIdx.x >> 6;
  const int u    = threadIdx.x & 63;
  const int atom = blockIdx.x * 4 + la;
  if (atom < nAtoms) {
    sx[la][u] = x [(size_t)atom * 64 + u];
    sh[la][u] = hp[(size_t)atom * 64 + u];
  }
  __syncthreads();
  if (atom >= nAtoms) return;

  float ir = bi[u], iz = bi[64 + u], inn = bi[128 + u];
  float hr = bh[u], hz = bh[64 + u], hn  = bh[128 + u];
  const float* wr0 = wi + (size_t)u * 64;
  const float* wz0 = wi + (size_t)(64 + u) * 64;
  const float* wn0 = wi + (size_t)(128 + u) * 64;
  const float* vr0 = wh + (size_t)u * 64;
  const float* vz0 = wh + (size_t)(64 + u) * 64;
  const float* vn0 = wh + (size_t)(128 + u) * 64;
  #pragma unroll 8
  for (int j = 0; j < 64; ++j) {
    const float xv = sx[la][j], hv = sh[la][j];
    ir += wr0[j] * xv; iz += wz0[j] * xv; inn += wn0[j] * xv;
    hr += vr0[j] * hv; hz += vz0[j] * hv; hn  += vn0[j] * hv;
  }
  const float r = 1.f / (1.f + __expf(-(ir + hr)));
  const float z = 1.f / (1.f + __expf(-(iz + hz)));
  const float n = tanhf(inn + r * hn);
  ho[(size_t)atom * 64 + u] = (1.f - z) * n + z * sh[la][u];
}

// ---------------------------------------------------------------------------
// Host side
// ---------------------------------------------------------------------------
static inline size_t alignUp(size_t v, size_t a) { return (v + a - 1) & ~(a - 1); }

template<int K>
static void launch_msg(const _Float16* xh, const float* feat, const int* idx,
                       const _Float16* Wb, float* out, int E, hipStream_t stream) {
  const int tiles  = (E + 15) / 16;
  int blocks = (tiles + NW - 1) / NW;
  if (blocks > 1024) blocks = 1024;
  const size_t smem = (size_t)(K + 1) * 64 * 64 * 2   // pre-swizzled weights
                    + (size_t)NW * 16 * (K + 1) * 2   // f16 features + sentinel
                    + (size_t)NW * 1024 * 2;          // x rows
  hipLaunchKernelGGL((msg_kernel<K>), dim3(blocks), dim3(TPB_MSG), smem, stream,
                     xh, feat, idx, Wb, out, E);
}

extern "C" void kernel_launch(void* const* d_in, const int* in_sizes, int n_in,
                              void* d_out, int out_size, void* d_ws, size_t ws_size,
                              hipStream_t stream) {
  (void)n_in; (void)out_size; (void)ws_size;
  const float* atomF = (const float*)d_in[0];
  const float* bondF = (const float*)d_in[1];
  const int*   bondI = (const int*)  d_in[2];
  const float* angF  = (const float*)d_in[3];
  const int*   angI  = (const int*)  d_in[4];
  const float* dihF  = (const float*)d_in[5];
  const int*   dihI  = (const int*)  d_in[6];
  const float* We = (const float*)d_in[7];  const float* be = (const float*)d_in[8];
  const float* Wa = (const float*)d_in[9];  const float* ba = (const float*)d_in[10];
  const float* Wd = (const float*)d_in[11]; const float* bd = (const float*)d_in[12];
  const float* gwi = (const float*)d_in[13];
  const float* gwh = (const float*)d_in[14];
  const float* gbi = (const float*)d_in[15];
  const float* gbh = (const float*)d_in[16];

  const int NA = in_sizes[0] / 64;
  const int Eb = in_sizes[2] / 2;
  const int Ea = in_sizes[4] / 2;
  const int Ed = in_sizes[6] / 2;
  const size_t rowBytes = (size_t)NA * 64 * sizeof(float);

  // Workspace carve-out
  char* ws = (char*)d_ws;
  size_t off = 0;
  auto carve = [&](size_t bytes) { void* p = ws + off; off = alignUp(off + bytes, 256); return p; };
  float*    h1  = (float*)carve(rowBytes);
  float*    h2  = (float*)carve(rowBytes);
  float*    mA  = (float*)carve(rowBytes);
  float*    mB  = (float*)carve(rowBytes);
  _Float16* xhF = (_Float16*)carve((size_t)NA * 64 * 2);
  _Float16* Wbe = (_Float16*)carve((size_t)17 * 4096 * 2);
  _Float16* Wba = (_Float16*)carve((size_t)9  * 4096 * 2);
  _Float16* Wbd = (_Float16*)carve((size_t)13 * 4096 * 2);

  // Build fragment-ordered f16 fused weights once per call.
  build_wbigA_kernel<16><<<(17 * 4096 + 255) / 256, 256, 0, stream>>>(We, be, Wbe);
  build_wbigA_kernel<8> <<<(9  * 4096 + 255) / 256, 256, 0, stream>>>(Wa, ba, Wba);
  build_wbigA_kernel<12><<<(13 * 4096 + 255) / 256, 256, 0, stream>>>(Wd, bd, Wbd);

  const int nElem = NA * 64;
  const int cvtBlocks = (nElem + 255) / 256;

  const float* hprev = atomF;
  for (int s = 0; s < 4; ++s) {
    float* hnext = (s == 3) ? (float*)d_out : ((s & 1) ? h2 : h1);

    // message 1: bonds (K=16), input = hprev
    f32_to_f16_kernel<<<cvtBlocks, 256, 0, stream>>>(hprev, xhF, nElem);
    hipMemsetAsync(mA, 0, rowBytes, stream);
    launch_msg<16>(xhF, bondF, bondI, Wbe, mA, Eb, stream);

    // message 2: angles (K=8), input = mA
    f32_to_f16_kernel<<<cvtBlocks, 256, 0, stream>>>(mA, xhF, nElem);
    hipMemsetAsync(mB, 0, rowBytes, stream);
    launch_msg<8>(xhF, angF, angI, Wba, mB, Ea, stream);

    // message 3: dihedrals (K=12), input = mB
    f32_to_f16_kernel<<<cvtBlocks, 256, 0, stream>>>(mB, xhF, nElem);
    hipMemsetAsync(mA, 0, rowBytes, stream);
    launch_msg<12>(xhF, dihF, dihI, Wbd, mA, Ed, stream);

    // GRU update
    gru_kernel<<<(NA + 3) / 4, 256, 0, stream>>>(mA, hprev, gwi, gwh, gbi, gbh,
                                                 hnext, NA);
    hprev = hnext;
  }
}
